// ContinuousRecurrentCore_87514253623306
// MI455X (gfx1250) — matrix operand order repaired and checked by reading, not compiled, not run on hardware
//
#include <hip/hip_runtime.h>
#include <hip/hip_bf16.h>

typedef __attribute__((ext_vector_type(16))) __bf16 v16bf;
typedef __attribute__((ext_vector_type(8)))  __bf16 v8bf;
typedef __attribute__((ext_vector_type(8)))  float  v8f;

#define H_DIM    1024
#define M_TOTAL  8192              // B*S = 8*1024
#define NLAYERS  4
#define MAXT     50
#define DT_C     0.05f
#define EPS_TAU  1e-6f
#define EPS_CONV 1e-3f

static const size_t MH = (size_t)M_TOTAL * H_DIM;          // 8M elements per layer-state

struct Ctl { unsigned int max_bits; int done; int t_conv; };

// ---- workspace layout (bytes) --------------------------------------------
static const size_t XB_OFF  = 0;                            // x bf16:       8M*2  = 16 MB
static const size_t WST_OFF = XB_OFF  + MH * 2;             // w_state bf16: 4M*2  =  8 MB
static const size_t WIN_OFF = WST_OFF + (size_t)NLAYERS * H_DIM * H_DIM * 2;
static const size_t WTA_OFF = WIN_OFF + (size_t)NLAYERS * H_DIM * H_DIM * 2;   // w_tau bf16: 8M*2
static const size_t S32_OFF = WTA_OFF + (size_t)NLAYERS * H_DIM * 2 * H_DIM * 2;
static const size_t S16_OFF = S32_OFF + (size_t)NLAYERS * MH * 4;              // 2 ping-pong bf16 state sets
static const size_t CTL_OFF = S16_OFF + 2 * (size_t)NLAYERS * MH * 2;

// ---- helpers --------------------------------------------------------------

__device__ __forceinline__ v8f wmma_bf16(v16bf a, v16bf b, v8f c) {
    // D(f32 16x16) = A(16x32 bf16) * B(32x16 bf16) + C
    return __builtin_amdgcn_wmma_f32_16x16x32_bf16(false, a, false, b, (short)0, c, false, false);
}

// A fragment: 16(M) x 32(K) bf16, row-major source with leading dim ld.
// ISA layout: lanes 0-15 (M=lane) hold K = {0..7, 16..23}; lanes 16-31 hold K = {8..15, 24..31}.
__device__ __forceinline__ v16bf load_a(const __bf16* __restrict__ p, int mrow, int kc, int lane, int ld) {
    const __bf16* base = p + (size_t)mrow * ld + kc + ((lane & 16) ? 8 : 0);
    union { v16bf v; v8bf h[2]; } u;
    u.h[0] = *(const v8bf*)(base);        // K offsets off .. off+7
    u.h[1] = *(const v8bf*)(base + 16);   // K offsets 16+off .. 23+off
    return u.v;
}

// B fragment: 32(K) x 16(N) bf16 where B[k][n] = W[nrow+n][kcol+k] (W row-major, ld = row length).
// ISA layout: lane = column (mod 16); lanes 0-15 hold K = {0..7, 16..23}; lanes 16-31 hold K = {8..15, 24..31}.
__device__ __forceinline__ v16bf load_b(const __bf16* __restrict__ W, int ld, int nrow, int kcol, int lane) {
    const __bf16* base = W + (size_t)nrow * ld + kcol + ((lane & 16) ? 8 : 0);
    union { v16bf v; v8bf h[2]; } u;
    u.h[0] = *(const v8bf*)(base);
    u.h[1] = *(const v8bf*)(base + 16);
    return u.v;
}

// ---- kernels ---------------------------------------------------------------

__global__ void cvt_bf16(const float* __restrict__ src, __bf16* __restrict__ dst, int n) {
    int i = blockIdx.x * blockDim.x + threadIdx.x;
    int stride = gridDim.x * blockDim.x;
    for (; i < n; i += stride) dst[i] = (__bf16)src[i];
}

__global__ void init_ctl(Ctl* ctl) {
    ctl->max_bits = 0u;
    ctl->done = 0;
    ctl->t_conv = MAXT;
}

__global__ void conv_check(Ctl* ctl, int t) {
    if (!ctl->done) {
        float md = __uint_as_float(ctl->max_bits);
        if (md < EPS_CONV) { ctl->t_conv = t; ctl->done = 1; }
    }
    ctl->max_bits = 0u;   // reset for next step
}

__global__ void finalize(const float* __restrict__ s3, float* __restrict__ out,
                         const Ctl* __restrict__ ctl, int n) {
    int i = blockIdx.x * blockDim.x + threadIdx.x;
    int stride = gridDim.x * blockDim.x;
    for (; i < n; i += stride) out[i] = s3[i];
    if (blockIdx.x == 0 && threadIdx.x == 0) ((int*)out)[n] = ctl->t_conv;
}

// Fused layer update: tau/forcing GEMMs + liquid-time-constant state update.
// Block tile 128(M) x 64(N), 8 waves, each wave 32x32 = 2x2 WMMA fragments,
// two f32 accumulators per fragment (tau pre-act, forcing pre-act).
__global__ __launch_bounds__(256)
void liquid_layer(const __bf16* __restrict__ cur,   // [M,H] layer input (bf16)
                  const __bf16* __restrict__ hin,   // [M,H] old state (bf16)
                  float*        __restrict__ h32,   // [M,H] f32 master state (in/out)
                  __bf16*       __restrict__ hout,  // [M,H] new state (bf16, other ping buffer)
                  const __bf16* __restrict__ Wtau,  // [H, 2H]  rows: out col; cols 0..H-1 -> cur, H..2H-1 -> h
                  const __bf16* __restrict__ Wst,   // [H, H]   multiplies h
                  const __bf16* __restrict__ Win,   // [H, H]   multiplies cur
                  const float*  __restrict__ btau,
                  const float*  __restrict__ bst,
                  const float*  __restrict__ bin,
                  Ctl*          __restrict__ ctl) {
    if (ctl->done) return;    // converged earlier: states frozen, nothing to do

    const int tid  = threadIdx.x;
    const int lane = tid & 31;
    const int w    = tid >> 5;
    const int wm   = w & 3;          // 4 wave-rows
    const int wn   = w >> 2;         // 2 wave-cols
    const int mb   = blockIdx.y * 128 + wm * 32;
    const int nb   = blockIdx.x * 64  + wn * 32;

    const v8f vzero = {0.f, 0.f, 0.f, 0.f, 0.f, 0.f, 0.f, 0.f};
    v8f acc1[2][2], acc2[2][2];
#pragma unroll
    for (int mf = 0; mf < 2; ++mf)
#pragma unroll
        for (int nf = 0; nf < 2; ++nf) { acc1[mf][nf] = vzero; acc2[mf][nf] = vzero; }

    for (int kc = 0; kc < H_DIM; kc += 32) {
        v16bf Ac[2], Ah[2];
#pragma unroll
        for (int mf = 0; mf < 2; ++mf) {
            const int m = mb + mf * 16 + (lane & 15);
            Ac[mf] = load_a(cur, m, kc, lane, H_DIM);
            Ah[mf] = load_a(hin, m, kc, lane, H_DIM);
        }
        v16bf Btx[2], Bth[2], Bs[2], Bi[2];
#pragma unroll
        for (int nf = 0; nf < 2; ++nf) {
            const int nrow = nb + nf * 16 + (lane & 15);
            Btx[nf] = load_b(Wtau, 2 * H_DIM, nrow, kc,          lane);
            Bth[nf] = load_b(Wtau, 2 * H_DIM, nrow, H_DIM + kc,  lane);
            Bs[nf]  = load_b(Wst,  H_DIM,     nrow, kc,          lane);
            Bi[nf]  = load_b(Win,  H_DIM,     nrow, kc,          lane);
        }
#pragma unroll
        for (int mf = 0; mf < 2; ++mf)
#pragma unroll
            for (int nf = 0; nf < 2; ++nf) {
                acc1[mf][nf] = wmma_bf16(Ac[mf], Btx[nf], acc1[mf][nf]);
                acc1[mf][nf] = wmma_bf16(Ah[mf], Bth[nf], acc1[mf][nf]);
                acc2[mf][nf] = wmma_bf16(Ah[mf], Bs[nf],  acc2[mf][nf]);
                acc2[mf][nf] = wmma_bf16(Ac[mf], Bi[nf],  acc2[mf][nf]);
            }
    }

    // Epilogue. C layout (f32 16x16): VGPR r: lanes 0-15 -> M=r, lanes 16-31 -> M=8+r; N = lane%16.
    float lmax = 0.f;
#pragma unroll
    for (int mf = 0; mf < 2; ++mf) {
#pragma unroll
        for (int nf = 0; nf < 2; ++nf) {
            const int n     = nb + nf * 16 + (lane & 15);
            const int mrow0 = mb + mf * 16 + ((lane & 16) ? 8 : 0);
            const float bt  = btau[n];
            const float bff = bst[n] + bin[n];
            v8f a1 = acc1[mf][nf];
            v8f a2 = acc2[mf][nf];
#pragma unroll
            for (int r = 0; r < 8; ++r) {
                const size_t idx = (size_t)(mrow0 + r) * H_DIM + n;
                const float tau  = 0.5f * (tanhf((a1[r] + bt) * 0.5f) + 1.0f); // stable sigmoid
                const float forc = tanhf(a2[r] + bff);
                const float hold = h32[idx];
                float hn = hold + DT_C * (-hold / (tau + EPS_TAU) + forc);
                hn = fminf(10.0f, fmaxf(-10.0f, hn));
                h32[idx]  = hn;
                hout[idx] = (__bf16)hn;
                lmax = fmaxf(lmax, fabsf(hn - hold));
            }
        }
    }

    __shared__ float red[256];
    red[tid] = lmax;
    __syncthreads();
#pragma unroll
    for (int s = 128; s > 0; s >>= 1) {
        if (tid < s) red[tid] = fmaxf(red[tid], red[tid + s]);
        __syncthreads();
    }
    if (tid == 0) atomicMax((unsigned int*)&ctl->max_bits, __float_as_uint(red[0]));
}

// ---- host launch -----------------------------------------------------------

extern "C" void kernel_launch(void* const* d_in, const int* in_sizes, int n_in,
                              void* d_out, int out_size, void* d_ws, size_t ws_size,
                              hipStream_t stream) {
    const float* x       = (const float*)d_in[0];
    const float* w_state = (const float*)d_in[1];
    const float* b_state = (const float*)d_in[2];
    const float* w_input = (const float*)d_in[3];
    const float* b_input = (const float*)d_in[4];
    const float* w_tau   = (const float*)d_in[5];
    const float* b_tau   = (const float*)d_in[6];

    char* ws = (char*)d_ws;
    __bf16* xb   = (__bf16*)(ws + XB_OFF);
    __bf16* wstb = (__bf16*)(ws + WST_OFF);
    __bf16* winb = (__bf16*)(ws + WIN_OFF);
    __bf16* wtab = (__bf16*)(ws + WTA_OFF);
    float*  s32  = (float*) (ws + S32_OFF);
    __bf16* s16  = (__bf16*)(ws + S16_OFF);   // two ping-pong sets of [NLAYERS][M,H]
    Ctl*    ctl  = (Ctl*)   (ws + CTL_OFF);

    const size_t pingElems = (size_t)NLAYERS * MH;

    // One-time (per call) conversions to bf16
    cvt_bf16<<<4096, 256, 0, stream>>>(x,       xb,   (int)MH);
    cvt_bf16<<<4096, 256, 0, stream>>>(w_state, wstb, NLAYERS * H_DIM * H_DIM);
    cvt_bf16<<<4096, 256, 0, stream>>>(w_input, winb, NLAYERS * H_DIM * H_DIM);
    cvt_bf16<<<4096, 256, 0, stream>>>(w_tau,   wtab, NLAYERS * H_DIM * 2 * H_DIM);

    hipMemsetAsync(s32, 0, (size_t)NLAYERS * MH * sizeof(float), stream);
    hipMemsetAsync(s16, 0, 2 * pingElems * sizeof(__bf16), stream);
    init_ctl<<<1, 1, 0, stream>>>(ctl);

    dim3 grid(H_DIM / 64, M_TOTAL / 128);   // (16, 64)
    for (int t = 0; t < MAXT; ++t) {
        const int p = t & 1;
        __bf16* sin_  = s16 + (size_t)p * pingElems;
        __bf16* sout_ = s16 + (size_t)(1 - p) * pingElems;
        for (int i = 0; i < NLAYERS; ++i) {
            const __bf16* cur = (i == 0) ? xb : (sout_ + (size_t)(i - 1) * MH);
            liquid_layer<<<grid, 256, 0, stream>>>(
                cur,
                sin_  + (size_t)i * MH,
                s32   + (size_t)i * MH,
                sout_ + (size_t)i * MH,
                wtab + (size_t)i * H_DIM * 2 * H_DIM,
                wstb + (size_t)i * H_DIM * H_DIM,
                winb + (size_t)i * H_DIM * H_DIM,
                b_tau + i * H_DIM, b_state + i * H_DIM, b_input + i * H_DIM,
                ctl);
        }
        conv_check<<<1, 1, 0, stream>>>(ctl, t);
    }

    finalize<<<4096, 256, 0, stream>>>(s32 + (size_t)(NLAYERS - 1) * MH, (float*)d_out, ctl, (int)MH);
}
